// GNNAnomalyDetector_29910152249706
// MI455X (gfx1250) — compile-verified, matrix operand order checked by reading, not run on hardware
//
#include <hip/hip_runtime.h>

// ---------------------------------------------------------------------------
// GATv2 GNN anomaly detector, gfx1250 (MI455X), wave32 + WMMA f16->f32 + TDM.
// ---------------------------------------------------------------------------

typedef __attribute__((ext_vector_type(16))) _Float16 v16h;
typedef __attribute__((ext_vector_type(8)))  _Float16 v8h;
typedef __attribute__((ext_vector_type(8)))  float    v8f;
typedef __attribute__((ext_vector_type(4)))  float    v4f;
typedef unsigned int u32x4 __attribute__((ext_vector_type(4)));
typedef int          i32x4 __attribute__((ext_vector_type(4)));
typedef int          i32x8 __attribute__((ext_vector_type(8)));

#if defined(__AMDGCN__) && __has_builtin(__builtin_amdgcn_tensor_load_to_lds) && \
    __has_builtin(__builtin_amdgcn_s_wait_tensorcnt)
#define HAS_TDM 1
#else
#define HAS_TDM 0
#endif

#define WMMA_F16(a, b, c) \
  __builtin_amdgcn_wmma_f32_16x16x32_f16(false, (a), false, (b), (short)0, (c), false, false)

// ---- TDM: DMA a K x 64 f32 tile (row-major, contiguous) global -> LDS -----
#if HAS_TDM
__device__ __forceinline__ void tdm_load_b_tile(const float* gB, void* lds_dst, int K) {
  unsigned long long ga = (unsigned long long)(size_t)gB;
  unsigned lds = (unsigned)(size_t)lds_dst;
  // D# group0: count=1 (valid), lds_addr, global_addr[56:0], type=2 ("image")
  u32x4 g0 = { 1u, lds, (unsigned)(ga & 0xffffffffull),
               (unsigned)((ga >> 32) & 0x01ffffffull) | 0x80000000u };
  // D# group1: data_size=4B(code 2); tensor_dim0=64; tensor_dim1=K;
  //            tile_dim0=64; tile_dim1=K; tile_dim2=0; tensor_dim0_stride=64
  i32x8 g1 = { (int)(2u << 16),            // wg_mask=0 | data_size=2 @ [17:16]
               (int)(64u << 16),           // tensor_dim0[15:0] @ bits[63:48 of desc]
               (int)(((unsigned)K) << 16), // tensor_dim0 hi=0 | tensor_dim1 lo=K
               (int)(64u << 16),           // tensor_dim1 hi=0 | tile_dim0=64
               (int)K,                     // tile_dim1=K | tile_dim2=0
               64, 0, 0 };                 // tensor_dim0_stride=64
  i32x4 z4 = { 0, 0, 0, 0 };
#if __clang_major__ >= 23
  i32x8 z8 = { 0, 0, 0, 0, 0, 0, 0, 0 };
  __builtin_amdgcn_tensor_load_to_lds(g0, g1, z4, z4, z8, 0);
#else
  __builtin_amdgcn_tensor_load_to_lds(g0, g1, z4, z4, 0);
#endif
}
#endif

// ---- fragment loaders -----------------------------------------------------
// A fragment (16x32 f16): lane = row M (lane&15), hi = lane>>4.
// halves 0..7  <- K = k0 + hi*8 + i ; halves 8..15 <- K = k0 + 16 + hi*8 + i
__device__ __forceinline__ v16h a_from_f32row(const float* __restrict__ row, int k0, int hi) {
  v4f a0 = *(const v4f*)(row + k0 + hi * 8);
  v4f a1 = *(const v4f*)(row + k0 + hi * 8 + 4);
  v4f a2 = *(const v4f*)(row + k0 + 16 + hi * 8);
  v4f a3 = *(const v4f*)(row + k0 + 16 + hi * 8 + 4);
  v16h a;
#pragma unroll
  for (int i = 0; i < 4; ++i) {
    a[i]      = (_Float16)a0[i];
    a[4 + i]  = (_Float16)a1[i];
    a[8 + i]  = (_Float16)a2[i];
    a[12 + i] = (_Float16)a3[i];
  }
  return a;
}

// B fragment from a K-contiguous (transposed, [n][k]) f16 row. Same K map as A.
__device__ __forceinline__ v16h frag_from_f16row(const _Float16* __restrict__ rowT, int k0, int hi) {
  v8h b0 = *(const v8h*)(rowT + k0 + hi * 8);
  v8h b1 = *(const v8h*)(rowT + k0 + 16 + hi * 8);
  v16h b;
#pragma unroll
  for (int i = 0; i < 8; ++i) { b[i] = b0[i]; b[8 + i] = b1[i]; }
  return b;
}

// ---- generic GEMM: C[M x 64] = A[M x K] @ B[K x 64] + bias ----------------
__global__ __launch_bounds__(256) void gemm_n64_kernel(
    const float* __restrict__ A, int lda, const float* __restrict__ B,
    const float* __restrict__ bias, float* __restrict__ C, int M, int K) {
  __shared__ _Float16 sB[64 * 72];  // B transposed: [n][k], pitch 72 halves
  const int tid = threadIdx.x;

#if HAS_TDM
  __shared__ float sBf[64 * 64];    // raw f32 weight tile, DMA'd by the TDM
  if (tid == 0) tdm_load_b_tile(B, (void*)sBf, K);
  if (tid < 32) __builtin_amdgcn_s_wait_tensorcnt(0);  // issuing wave drains TENSORcnt
  __syncthreads();
  for (int idx = tid; idx < K * 64; idx += 256) {
    int k = idx >> 6, n = idx & 63;
    sB[n * 72 + k] = (_Float16)sBf[idx];
  }
#else
  for (int idx = tid; idx < K * 64; idx += 256) {
    int k = idx >> 6, n = idx & 63;
    sB[n * 72 + k] = (_Float16)B[idx];
  }
#endif
  __syncthreads();

  const int wave = tid >> 5, lane = tid & 31;
  const int ln16 = lane & 15, hi = lane >> 4;
  const int m0 = blockIdx.x * 128 + wave * 16;
  int mr = m0 + ln16; if (mr > M - 1) mr = M - 1;
  const float* arow = A + (size_t)mr * lda;

  v8f acc[4];
#pragma unroll
  for (int t = 0; t < 4; ++t) {
    float bv = bias ? bias[t * 16 + ln16] : 0.f;
#pragma unroll
    for (int r = 0; r < 8; ++r) acc[t][r] = bv;
  }

  for (int k0 = 0; k0 < K; k0 += 32) {
    v16h a = a_from_f32row(arow, k0, hi);
#pragma unroll
    for (int t = 0; t < 4; ++t) {
      v16h b = frag_from_f16row(&sB[(t * 16 + ln16) * 72], k0, hi);
      acc[t] = WMMA_F16(a, b, acc[t]);
    }
  }

  // D layout: VGPR r of acc holds row (m0 + r + 8*hi), column t*16 + ln16.
  float* crow = C + (size_t)(m0 + 8 * hi) * 64;
  if (m0 + 16 <= M) {            // fast path: whole wave tile in range
#pragma unroll
    for (int t = 0; t < 4; ++t) {
      int n = t * 16 + ln16;
#pragma unroll
      for (int r = 0; r < 8; ++r) crow[r * 64 + n] = acc[t][r];
    }
  } else {                        // tail block only
#pragma unroll
    for (int t = 0; t < 4; ++t) {
      int n = t * 16 + ln16;
#pragma unroll
      for (int r = 0; r < 8; ++r) {
        if (m0 + r + 8 * hi < M) crow[r * 64 + n] = acc[t][r];
      }
    }
  }
}

// ---- small utility kernels ------------------------------------------------
__global__ void zero_kernel(float* p, int n) {
  int i = blockIdx.x * blockDim.x + threadIdx.x;
  if (i < n) p[i] = 0.f;
}

__global__ void ea_colsum_kernel(const float* __restrict__ ea, float* __restrict__ acc, int E) {
  __shared__ float s[256];
  int col = threadIdx.x & 31, rowset = threadIdx.x >> 5;
  float p = 0.f;
  for (int r = blockIdx.x * 8 + rowset; r < E; r += gridDim.x * 8)
    p += ea[(size_t)r * 32 + col];
  s[threadIdx.x] = p;
  __syncthreads();
  if (threadIdx.x < 32) {
    float t = 0.f;
#pragma unroll
    for (int w = 0; w < 8; ++w) t += s[w * 32 + col];
    unsafeAtomicAdd(&acc[col], t);
  }
}

// ewl[j] = mean(edge_attr) @ We[:, j]   (self-loop edge-feature transform)
__global__ void ewl_kernel(const float* __restrict__ eacc, const float* __restrict__ We,
                           float* __restrict__ ewl, float invE) {
  int j = threadIdx.x;  // 64
  float a = 0.f;
  for (int c = 0; c < 32; ++c) a += eacc[c] * invE * We[c * 64 + j];
  ewl[j] = a;
}

__global__ void transpose_f16_kernel(const float* __restrict__ W, _Float16* __restrict__ Wt,
                                     int K, int Nn) {
  int idx = blockIdx.x * blockDim.x + threadIdx.x;
  if (idx < K * Nn) {
    int k = idx / Nn, n = idx - k * Nn;
    Wt[n * K + k] = (_Float16)W[idx];
  }
}

__global__ void init_layer_kernel(float* out, float* smax, float* denom, int N, int H) {
  int i = blockIdx.x * blockDim.x + threadIdx.x;
  if (i < N * 64) out[i] = 0.f;
  if (i < N * H) { smax[i] = -3.0e38f; denom[i] = 0.f; }
}

// ---- GATv2 attention pipeline --------------------------------------------
__global__ void score_kernel(const float* __restrict__ xl, const float* __restrict__ xr,
                             const float* __restrict__ ew, const float* __restrict__ ewl,
                             const int* __restrict__ src, const int* __restrict__ dst,
                             const float* __restrict__ att, float* __restrict__ score,
                             int E, int Ntot, int H, int C) {
  int idx = blockIdx.x * blockDim.x + threadIdx.x;
  if (idx >= Ntot * H) return;
  int e = idx / H, h = idx - e * H;
  int s, d; const float* ep;
  if (e < E) { s = src[e]; d = dst[e]; ep = ew + (size_t)e * 64 + h * C; }
  else       { s = d = e - E;          ep = ewl + h * C; }
  const float* lp = xl + (size_t)s * 64 + h * C;
  const float* rp = xr + (size_t)d * 64 + h * C;
  const float* ap = att + h * C;
  float acc = 0.f;
  for (int c = 0; c < C; ++c) {
    float m = lp[c] + rp[c] + ep[c];
    m = m > 0.f ? m : 0.2f * m;  // leaky_relu(0.2)
    acc += m * ap[c];
  }
  score[idx] = acc;
}

__global__ void smax_kernel(const float* __restrict__ score, const int* __restrict__ dst,
                            float* __restrict__ smax, int E, int Ntot, int H) {
  int idx = blockIdx.x * blockDim.x + threadIdx.x;
  if (idx >= Ntot * H) return;
  int e = idx / H, h = idx - e * H;
  int d = (e < E) ? dst[e] : (e - E);
  unsafeAtomicMax(&smax[d * H + h], score[idx]);
}

__global__ void ex_kernel(float* __restrict__ score, const int* __restrict__ dst,
                          const float* __restrict__ smax, float* __restrict__ denom,
                          int E, int Ntot, int H) {
  int idx = blockIdx.x * blockDim.x + threadIdx.x;
  if (idx >= Ntot * H) return;
  int e = idx / H, h = idx - e * H;
  int d = (e < E) ? dst[e] : (e - E);
  float ex = __expf(score[idx] - smax[d * H + h]);
  score[idx] = ex;  // reuse buffer
  unsafeAtomicAdd(&denom[d * H + h], ex);
}

__global__ void agg_kernel(const float* __restrict__ ex, const float* __restrict__ denom,
                           const float* __restrict__ xl, const int* __restrict__ src,
                           const int* __restrict__ dst, float* __restrict__ out,
                           int E, int Ntot, int H, int C) {
  int idx = blockIdx.x * blockDim.x + threadIdx.x;
  if (idx >= Ntot * H) return;
  int e = idx / H, h = idx - e * H;
  int s, d;
  if (e < E) { s = src[e]; d = dst[e]; } else { s = d = e - E; }
  float alpha = ex[idx] / (denom[d * H + h] + 1e-16f);
  const float* lp = xl + (size_t)s * 64 + h * C;
  float* op = out + (size_t)d * 64 + h * C;
  for (int c = 0; c < C; ++c) unsafeAtomicAdd(&op[c], alpha * lp[c]);
}

// bias add + layernorm(64) + relu
__global__ void post_kernel(const float* __restrict__ out, const float* __restrict__ cbias,
                            const float* __restrict__ g, const float* __restrict__ b,
                            float* __restrict__ z, int N) {
  int i = blockIdx.x * blockDim.x + threadIdx.x;
  if (i >= N) return;
  const float* op = out + (size_t)i * 64;
  float v[64]; float mu = 0.f;
#pragma unroll
  for (int c = 0; c < 64; ++c) { v[c] = op[c] + cbias[c]; mu += v[c]; }
  mu *= (1.f / 64.f);
  float var = 0.f;
#pragma unroll
  for (int c = 0; c < 64; ++c) { float dd = v[c] - mu; var += dd * dd; }
  var *= (1.f / 64.f);
  float inv = rsqrtf(var + 1e-5f);
  float* zp = z + (size_t)i * 64;
#pragma unroll
  for (int c = 0; c < 64; ++c) {
    float y = (v[c] - mu) * inv * g[c] + b[c];
    zp[c] = y > 0.f ? y : 0.f;
  }
}

// ---- fused edge classifier: rep(224) -> 64 -> 32 -> 1, all in LDS ---------
__global__ __launch_bounds__(256) void clf_fused_kernel(
    const float* __restrict__ z, const int* __restrict__ src, const int* __restrict__ dst,
    const float* __restrict__ ea,
    const _Float16* __restrict__ w1t, const float* __restrict__ b1,
    const _Float16* __restrict__ w2t, const float* __restrict__ b2,
    const float* __restrict__ w3, const float* __restrict__ b3,
    float* __restrict__ out, int E) {
  extern __shared__ char smem[];
  _Float16* rep = (_Float16*)smem;       // 128 x 232 (pitch 464B, 16B-aligned)
  _Float16* sh1 = rep + 128 * 232;       // 128 x 72
  _Float16* sh2 = sh1 + 128 * 72;        // 128 x 40

  const int tid = threadIdx.x;
  // phase 0: build rep tile = [zs | zd | zs*zd | ea]
  {
    int el = tid & 127, part = tid >> 7;
    int ej = blockIdx.x * 128 + el; if (ej > E - 1) ej = E - 1;
    int s = src[ej], d = dst[ej];
    const float* zs = z + (size_t)s * 64;
    const float* zd = z + (size_t)d * 64;
    const float* ep = ea + (size_t)ej * 32;
    _Float16* rr = rep + el * 232;
    int c0 = part * 112;
    for (int c = c0; c < c0 + 112; ++c) {
      float v;
      if (c < 64)       v = zs[c];
      else if (c < 128) v = zd[c - 64];
      else if (c < 192) v = zs[c - 128] * zd[c - 128];
      else              v = ep[c - 192];
      rr[c] = (_Float16)v;
    }
  }
  __syncthreads();

  const int wave = tid >> 5, lane = tid & 31;
  const int ln16 = lane & 15, hi = lane >> 4;
  const int row = wave * 16 + ln16;

  // phase 1: h1 = relu(rep @ W1 + b1)  (K=224, N=64)
  {
    v8f acc[4];
#pragma unroll
    for (int t = 0; t < 4; ++t) {
      float bv = b1[t * 16 + ln16];
#pragma unroll
      for (int r = 0; r < 8; ++r) acc[t][r] = bv;
    }
    for (int k0 = 0; k0 < 224; k0 += 32) {
      v16h a = frag_from_f16row(rep + row * 232, k0, hi);
#pragma unroll
      for (int t = 0; t < 4; ++t) {
        v16h b = frag_from_f16row(w1t + (t * 16 + ln16) * 224, k0, hi);
        acc[t] = WMMA_F16(a, b, acc[t]);
      }
    }
#pragma unroll
    for (int t = 0; t < 4; ++t) {
      int n = t * 16 + ln16;
#pragma unroll
      for (int r = 0; r < 8; ++r) {
        int rr = wave * 16 + r + 8 * hi;
        float v = acc[t][r]; v = v > 0.f ? v : 0.f;
        sh1[rr * 72 + n] = (_Float16)v;
      }
    }
  }
  __syncthreads();

  // phase 2: h2 = relu(h1 @ W2 + b2)  (K=64, N=32)
  {
    v8f acc[2];
#pragma unroll
    for (int t = 0; t < 2; ++t) {
      float bv = b2[t * 16 + ln16];
#pragma unroll
      for (int r = 0; r < 8; ++r) acc[t][r] = bv;
    }
    for (int k0 = 0; k0 < 64; k0 += 32) {
      v16h a = frag_from_f16row(sh1 + row * 72, k0, hi);
#pragma unroll
      for (int t = 0; t < 2; ++t) {
        v16h b = frag_from_f16row(w2t + (t * 16 + ln16) * 64, k0, hi);
        acc[t] = WMMA_F16(a, b, acc[t]);
      }
    }
#pragma unroll
    for (int t = 0; t < 2; ++t) {
      int n = t * 16 + ln16;
#pragma unroll
      for (int r = 0; r < 8; ++r) {
        int rr = wave * 16 + r + 8 * hi;
        float v = acc[t][r]; v = v > 0.f ? v : 0.f;
        sh2[rr * 40 + n] = (_Float16)v;
      }
    }
  }
  __syncthreads();

  // phase 3: logits = h2 @ W3 + b3
  if (tid < 128) {
    int ej = blockIdx.x * 128 + tid;
    if (ej < E) {
      float acc = b3[0];
      for (int k = 0; k < 32; ++k) acc += (float)sh2[tid * 40 + k] * w3[k];
      out[ej] = acc;
    }
  }
}

// ---------------------------------------------------------------------------
extern "C" void kernel_launch(void* const* d_in, const int* in_sizes, int n_in,
                              void* d_out, int out_size, void* d_ws, size_t ws_size,
                              hipStream_t stream) {
  (void)in_sizes; (void)n_in; (void)out_size; (void)ws_size;
  const int N = 50000, E = 800000, NT = E + N;

  // input order: x, edge_index, edge_attr, then params pytree leaves
  // (clf: W1,W2,W3,b1,b2,b3; conv_l: We,Wl,Wr,att,bias,bl,br; norm_l: b,g)
  const float* x  = (const float*)d_in[0];
  const int*   ei = (const int*)  d_in[1];
  const float* ea = (const float*)d_in[2];
  const float* W1 = (const float*)d_in[3];
  const float* W2 = (const float*)d_in[4];
  const float* W3 = (const float*)d_in[5];
  const float* b1 = (const float*)d_in[6];
  const float* b2 = (const float*)d_in[7];
  const float* b3 = (const float*)d_in[8];
  const int* src = ei;
  const int* dst = ei + E;

  float* ws = (float*)d_ws;
  size_t o = 0;
  float* ew    = ws + o; o += (size_t)E * 64;
  float* xl    = ws + o; o += (size_t)N * 64;
  float* xr    = ws + o; o += (size_t)N * 64;
  float* aggo  = ws + o; o += (size_t)N * 64;
  float* zA    = ws + o; o += (size_t)N * 64;
  float* zB    = ws + o; o += (size_t)N * 64;
  float* score = ws + o; o += (size_t)NT * 4;
  float* smax  = ws + o; o += (size_t)N * 4;
  float* denom = ws + o; o += (size_t)N * 4;
  float* eacc  = ws + o; o += 32;
  float* ewl   = ws + o; o += 64;
  _Float16* w1t = (_Float16*)(ws + o); o += (224 * 64) / 2;
  _Float16* w2t = (_Float16*)(ws + o); o += (64 * 32) / 2;

  dim3 blk(256);

  // edge_attr column mean (shared by all layers' self-loop features)
  hipLaunchKernelGGL(zero_kernel, dim3(1), dim3(32), 0, stream, eacc, 32);
  hipLaunchKernelGGL(ea_colsum_kernel, dim3(512), blk, 0, stream, ea, eacc, E);
  // classifier weights -> f16 transposed
  hipLaunchKernelGGL(transpose_f16_kernel, dim3((224 * 64 + 255) / 256), blk, 0, stream, W1, w1t, 224, 64);
  hipLaunchKernelGGL(transpose_f16_kernel, dim3((64 * 32 + 255) / 256), blk, 0, stream, W2, w2t, 64, 32);

  const float* zin = x;
  float* zout = zA;
  for (int l = 0; l < 3; ++l) {
    const float* We  = (const float*)d_in[9 + 7 * l + 0];
    const float* Wl  = (const float*)d_in[9 + 7 * l + 1];
    const float* Wr  = (const float*)d_in[9 + 7 * l + 2];
    const float* att = (const float*)d_in[9 + 7 * l + 3];
    const float* cb  = (const float*)d_in[9 + 7 * l + 4];
    const float* bl  = (const float*)d_in[9 + 7 * l + 5];
    const float* br  = (const float*)d_in[9 + 7 * l + 6];
    const float* lnb = (const float*)d_in[30 + 2 * l + 0];
    const float* lng = (const float*)d_in[30 + 2 * l + 1];
    const int H = (l == 2) ? 1 : 4;
    const int C = 64 / H;
    const int nsh = NT * H;

    hipLaunchKernelGGL(gemm_n64_kernel, dim3((N + 127) / 128), blk, 0, stream,
                       zin, 64, Wl, bl, xl, N, 64);
    hipLaunchKernelGGL(gemm_n64_kernel, dim3((N + 127) / 128), blk, 0, stream,
                       zin, 64, Wr, br, xr, N, 64);
    hipLaunchKernelGGL(gemm_n64_kernel, dim3((E + 127) / 128), blk, 0, stream,
                       ea, 32, We, (const float*)nullptr, ew, E, 32);
    hipLaunchKernelGGL(ewl_kernel, dim3(1), dim3(64), 0, stream, eacc, We, ewl, 1.f / (float)E);
    hipLaunchKernelGGL(init_layer_kernel, dim3((N * 64 + 255) / 256), blk, 0, stream,
                       aggo, smax, denom, N, H);
    hipLaunchKernelGGL(score_kernel, dim3((nsh + 255) / 256), blk, 0, stream,
                       xl, xr, ew, ewl, src, dst, att, score, E, NT, H, C);
    hipLaunchKernelGGL(smax_kernel, dim3((nsh + 255) / 256), blk, 0, stream,
                       score, dst, smax, E, NT, H);
    hipLaunchKernelGGL(ex_kernel, dim3((nsh + 255) / 256), blk, 0, stream,
                       score, dst, smax, denom, E, NT, H);
    hipLaunchKernelGGL(agg_kernel, dim3((nsh + 255) / 256), blk, 0, stream,
                       score, denom, xl, src, dst, aggo, E, NT, H, C);
    hipLaunchKernelGGL(post_kernel, dim3((N + 255) / 256), blk, 0, stream,
                       aggo, cb, lng, lnb, zout, N);

    zin = zout;
    zout = (l == 0) ? zB : zA;
  }

  size_t shmem = (size_t)(128 * 232 + 128 * 72 + 128 * 40) * sizeof(_Float16);  // 88 KB
  hipLaunchKernelGGL(clf_fused_kernel, dim3((E + 127) / 128), dim3(256), shmem, stream,
                     zin, src, dst, ea, w1t, b1, w2t, b2, W3, b3, (float*)d_out, E);
}
